// gmm_hyper_y1_34565896798757
// MI455X (gfx1250) — matrix operand order repaired and checked by reading, not compiled, not run on hardware
//
#include <hip/hip_runtime.h>
#include <hip/hip_bf16.h>

typedef __bf16 bf16_t;
typedef __attribute__((ext_vector_type(16))) __bf16 v16bf;
typedef __attribute__((ext_vector_type(8)))  __bf16 v8bf;
typedef __attribute__((ext_vector_type(8)))  float   v8f;

// ---------------- sizes ----------------
#define NB   16          // batch
#define NC   192         // N channels (also Cin everywhere)
#define MKC  576         // M*K channels
#define KTAP 25
#define KTOT (NC * KTAP) // 4800 = 150 * 32
#define NSTEP 150        // K-steps of 32 (25 taps x 6 ci-blocks)
#define SIG_ELEMS (16ull * 576 * 64 * 64)   // 37,748,736

// padded/dilated buffer spatial dims (channel-last: [b][y][x][NC])
#define ZH 36   // stage0: 16 -> dilated 31 + pads (2,3)
#define PH 68   // stage1/2: 32 -> dilated 63 + pads (2,3) ; or 64 + 2 + 2

// ---------------- CDNA5 async global->LDS helpers ----------------
__device__ __forceinline__ void async_b128(unsigned lds_off, unsigned long long gaddr) {
    asm volatile("global_load_async_to_lds_b128 %0, %1, off"
                 :: "v"(lds_off), "v"(gaddr) : "memory");
}
__device__ __forceinline__ void wait_async0() {
    asm volatile("s_wait_asynccnt 0x0" ::: "memory");
}

// ---------------- pack kernels ----------------
// z1 (B,NC,16,16) fp32 -> Z channel-last dilated+padded (B,ZH,ZH,NC) bf16
__global__ __launch_bounds__(256) void pack_z_kernel(const float* __restrict__ z,
                                                     bf16_t* __restrict__ Z) {
    size_t i = (size_t)blockIdx.x * 256 + threadIdx.x;
    const size_t total = (size_t)NB * ZH * ZH * NC;
    if (i >= total) return;
    int c = (int)(i % NC); size_t r = i / NC;
    int x = (int)(r % ZH); r /= ZH;
    int y = (int)(r % ZH); int b = (int)(r / ZH);
    float v = 0.0f;
    int dy = y - 2, dx = x - 2;
    if (dy >= 0 && dx >= 0 && !(dy & 1) && !(dx & 1) && (dy >> 1) < 16 && (dx >> 1) < 16)
        v = z[(((size_t)b * NC + c) * 16 + (dy >> 1)) * 16 + (dx >> 1)];
    Z[i] = (bf16_t)v;
}

// deconv weights (Cin,Cout,5,5) -> A[co][tap*Cin + ci] = w[ci][co][4-ky][4-kx]
__global__ __launch_bounds__(256) void pack_wdec_kernel(const float* __restrict__ w,
                                                        bf16_t* __restrict__ A,
                                                        int Cin, int Cout) {
    size_t i = (size_t)blockIdx.x * 256 + threadIdx.x;
    size_t total = (size_t)Cout * Cin * KTAP;
    if (i >= total) return;
    int co  = (int)(i / (size_t)(Cin * KTAP));
    int r   = (int)(i % (size_t)(Cin * KTAP));
    int tap = r / Cin, ci = r % Cin;
    int ky = tap / 5, kx = tap % 5;
    A[i] = (bf16_t)w[(((size_t)ci * Cout + co) * 5 + (4 - ky)) * 5 + (4 - kx)];
}

// conv weights (Cout,Cin,5,5) -> A[co][tap*Cin + ci] = w[co][ci][ky][kx]
__global__ __launch_bounds__(256) void pack_wconv_kernel(const float* __restrict__ w,
                                                         bf16_t* __restrict__ A,
                                                         int Cin, int Cout) {
    size_t i = (size_t)blockIdx.x * 256 + threadIdx.x;
    size_t total = (size_t)Cout * Cin * KTAP;
    if (i >= total) return;
    int co  = (int)(i / (size_t)(Cin * KTAP));
    int r   = (int)(i % (size_t)(Cin * KTAP));
    int tap = r / Cin, ci = r % Cin;
    int ky = tap / 5, kx = tap % 5;
    A[i] = (bf16_t)w[(((size_t)co * Cin + ci) * 5 + ky) * 5 + kx];
}

// ---------------- core implicit-GEMM conv (k5) via WMMA bf16 ----------------
// Input is channel-last padded (b, Hp, Wp, NC). K-dim of each WMMA step = 32 input
// channels of one tap -> all staging is contiguous 16B async loads to LDS.
// ACT: 0 none, 1 relu, 2 leaky
// MODE: 0 dilated-strided bf16 (ch-last), 1 pad2 bf16 (ch-last), 2 fp32 NCHW out,
//       3 global-max reduce (ordered-uint atomicMax)
template <int ACT, int MODE>
__global__ __launch_bounds__(256)
void conv5_gemm(const bf16_t* __restrict__ Xp, int Hp, int Wp,
                const bf16_t* __restrict__ Aw, const float* __restrict__ bias,
                int Cout, int wsh /*Wout=1<<wsh*/,
                void* __restrict__ dst, int Hd, int Wd) {
    const int Wout = 1 << wsh;
    const int t    = threadIdx.x;
    const int lane = t & 31, wave = t >> 5;
    const int wm = wave >> 2, wn = wave & 3;     // 2 x 4 waves of 32x32
    const int m0 = blockIdx.y * 64;
    const int n0 = blockIdx.x * 128;
    const int b  = blockIdx.z;

    // double-buffered LDS; rows padded to 40 elems (80B) -> conflict-free b128
    __shared__ __align__(16) bf16_t lsA[2][64 * 40];    // [m][32ci]
    __shared__ __align__(16) bf16_t lsB[2][128 * 40];   // [pixel][32ci]

    v8f acc[2][2] = {};

    const bf16_t* Xb = Xp + (size_t)b * Hp * Wp * NC;

    // per-thread staging coords
    const int pl   = t & 127;                    // B: pixel in tile
    const int seg  = t >> 7;                     // B: ci half (0/1)
    const int pg   = n0 + pl;
    const int pbase = (pg >> wsh) * Wp + (pg & (Wout - 1));  // py*Wp+px
    const int arow = t >> 2;                     // A: row in tile
    const int achk = (t & 3) << 3;               // A: 8-elem chunk

    const unsigned lsA_base = (unsigned)(uintptr_t)&lsA[0][0];
    const unsigned lsB_base = (unsigned)(uintptr_t)&lsB[0][0];
    const unsigned lsA_off  = (unsigned)((arow * 40 + achk) * 2);
    const unsigned lsB_off  = (unsigned)((pl * 40 + seg * 16) * 2);

    const bf16_t* aptr = Aw + (size_t)(m0 + arow) * KTOT + achk;

    // issue staging for step s into buffer buf
    auto issue = [&](int s, int buf) {
        int tap = s / 6;                      // uniform
        int ci0 = (s - tap * 6) * 32;         // uniform
        int ky  = tap / 5, kx = tap - ky * 5; // uniform
        int u   = ky * Wp + kx;               // uniform
        unsigned ab = lsA_base + (unsigned)buf * (64 * 40 * 2) + lsA_off;
        unsigned bb = lsB_base + (unsigned)buf * (128 * 40 * 2) + lsB_off;
        async_b128(ab, (unsigned long long)(uintptr_t)(aptr + tap * NC + ci0));
        const bf16_t* bsrc = Xb + (size_t)(pbase + u) * NC + ci0 + seg * 16;
        async_b128(bb,      (unsigned long long)(uintptr_t)bsrc);
        async_b128(bb + 16, (unsigned long long)(uintptr_t)(bsrc + 8));
    };

    const int kh = lane >> 4;   // K-half selector
    const int ml = lane & 15;

    issue(0, 0);
    for (int s = 0; s < NSTEP; ++s) {
        const int buf = s & 1;
        wait_async0();
        __syncthreads();

        // fragments: two contiguous 16B LDS reads each
        const bf16_t* pa0 = &lsA[buf][(wm * 32 + ml) * 40];
        v8bf a0lo = *(const v8bf*)(pa0 + kh * 8);
        v8bf a0hi = *(const v8bf*)(pa0 + 16 + kh * 8);
        v16bf a0 = __builtin_shufflevector(a0lo, a0hi, 0,1,2,3,4,5,6,7,8,9,10,11,12,13,14,15);
        const bf16_t* pa1 = &lsA[buf][(wm * 32 + 16 + ml) * 40];
        v8bf a1lo = *(const v8bf*)(pa1 + kh * 8);
        v8bf a1hi = *(const v8bf*)(pa1 + 16 + kh * 8);
        v16bf a1 = __builtin_shufflevector(a1lo, a1hi, 0,1,2,3,4,5,6,7,8,9,10,11,12,13,14,15);
        const bf16_t* pb0 = &lsB[buf][(wn * 32 + ml) * 40];
        v8bf b0lo = *(const v8bf*)(pb0 + kh * 8);
        v8bf b0hi = *(const v8bf*)(pb0 + 16 + kh * 8);
        v16bf b0 = __builtin_shufflevector(b0lo, b0hi, 0,1,2,3,4,5,6,7,8,9,10,11,12,13,14,15);
        const bf16_t* pb1 = &lsB[buf][(wn * 32 + 16 + ml) * 40];
        v8bf b1lo = *(const v8bf*)(pb1 + kh * 8);
        v8bf b1hi = *(const v8bf*)(pb1 + 16 + kh * 8);
        v16bf b1 = __builtin_shufflevector(b1lo, b1hi, 0,1,2,3,4,5,6,7,8,9,10,11,12,13,14,15);

        if (s + 1 < NSTEP) issue(s + 1, buf ^ 1);   // overlap with WMMA

        acc[0][0] = __builtin_amdgcn_wmma_f32_16x16x32_bf16(false, a0, false, b0, (short)0, acc[0][0], false, false);
        acc[0][1] = __builtin_amdgcn_wmma_f32_16x16x32_bf16(false, a0, false, b1, (short)0, acc[0][1], false, false);
        acc[1][0] = __builtin_amdgcn_wmma_f32_16x16x32_bf16(false, a1, false, b0, (short)0, acc[1][0], false, false);
        acc[1][1] = __builtin_amdgcn_wmma_f32_16x16x32_bf16(false, a1, false, b1, (short)0, acc[1][1], false, false);
    }

    // --- epilogue: bias + activation + route ---
    const int colg = lane & 15;
    const int mh   = (lane >> 4) * 8;   // C/D layout: lanes 16-31 hold M = r+8
#pragma unroll
    for (int fm = 0; fm < 2; ++fm)
#pragma unroll
    for (int fn = 0; fn < 2; ++fn) {
#pragma unroll
        for (int r = 0; r < 8; ++r) {
            int co = m0 + wm * 32 + fm * 16 + mh + r;
            int p  = n0 + wn * 32 + fn * 16 + colg;
            float v = acc[fm][fn][r] + bias[co];
            if (ACT == 1) v = v > 0.0f ? v : 0.0f;
            if (ACT == 2) v = v >= 0.0f ? v : 0.01f * v;
            int y = p >> wsh, x = p & (Wout - 1);
            if (MODE == 0) {        // channel-last, dilation-2 + pad-2 position
                ((bf16_t*)dst)[(((size_t)b * Hd + 2 * y + 2) * Wd + 2 * x + 2) * Cout + co] = (bf16_t)v;
            } else if (MODE == 1) { // channel-last, pad-2 position
                ((bf16_t*)dst)[(((size_t)b * Hd + y + 2) * Wd + x + 2) * Cout + co] = (bf16_t)v;
            } else if (MODE == 2) { // NCHW fp32
                ((float*)dst)[(((size_t)b * Cout + co) << (2 * wsh)) + p] = v;
            } else {                // MODE 3: max over pixels
                float m = v;
#pragma unroll
                for (int sft = 8; sft >= 1; sft >>= 1) {
                    float o = __shfl_xor(m, sft, 32);
                    m = fmaxf(m, o);
                }
                if (colg == 0) {
                    unsigned bits = __float_as_uint(m);
                    unsigned key  = (bits & 0x80000000u) ? ~bits : (bits | 0x80000000u);
                    atomicMax((unsigned*)dst + (size_t)b * Cout + co, key);
                }
            }
        }
    }
}

// ---------------- weights tail: leaky -> 576x576 matvec -> softmax over K=3 ----------------
__global__ __launch_bounds__(256)
void weights_tail_kernel(const unsigned* __restrict__ Umax, const float* __restrict__ w3,
                         const float* __restrict__ b3, float* __restrict__ out) {
    __shared__ float v[MKC];
    __shared__ float o[MKC];
    const int b = blockIdx.x, t = threadIdx.x;
    for (int i = t; i < MKC; i += 256) {
        unsigned u    = Umax[b * MKC + i];
        unsigned bits = (u & 0x80000000u) ? (u ^ 0x80000000u) : ~u;
        float f = __uint_as_float(bits);
        v[i] = f >= 0.0f ? f : 0.01f * f;
    }
    __syncthreads();
    for (int co = t; co < MKC; co += 256) {
        float s = b3[co];
        for (int ci = 0; ci < MKC; ++ci) s += w3[(size_t)co * MKC + ci] * v[ci];
        o[co] = s;
    }
    __syncthreads();
    if (t < 192) {
        float a0 = o[t], a1 = o[192 + t], a2 = o[384 + t];
        float mx = fmaxf(a0, fmaxf(a1, a2));
        float e0 = __expf(a0 - mx), e1 = __expf(a1 - mx), e2 = __expf(a2 - mx);
        float inv = 1.0f / (e0 + e1 + e2);
        out[(size_t)b * MKC + t]       = e0 * inv;
        out[(size_t)b * MKC + 192 + t] = e1 * inv;
        out[(size_t)b * MKC + 384 + t] = e2 * inv;
    }
}

// ---------------- launch ----------------
extern "C" void kernel_launch(void* const* d_in, const int* in_sizes, int n_in,
                              void* d_out, int out_size, void* d_ws, size_t ws_size,
                              hipStream_t stream) {
    (void)in_sizes; (void)n_in; (void)out_size; (void)ws_size;
    const float* z1    = (const float*)d_in[0];
    const float* gs_w1 = (const float*)d_in[1];  const float* gs_b1 = (const float*)d_in[2];
    const float* gs_w2 = (const float*)d_in[3];  const float* gs_b2 = (const float*)d_in[4];
    const float* gs_w3 = (const float*)d_in[5];  const float* gs_b3 = (const float*)d_in[6];
    const float* gm_w1 = (const float*)d_in[7];  const float* gm_b1 = (const float*)d_in[8];
    const float* gm_w2 = (const float*)d_in[9];  const float* gm_b2 = (const float*)d_in[10];
    const float* gm_w3 = (const float*)d_in[11]; const float* gm_b3 = (const float*)d_in[12];
    const float* gw_w1 = (const float*)d_in[13]; const float* gw_b1 = (const float*)d_in[14];
    const float* gw_w2 = (const float*)d_in[15]; const float* gw_b2 = (const float*)d_in[16];
    const float* gw_w3 = (const float*)d_in[17]; const float* gw_b3 = (const float*)d_in[18];
    float* out = (float*)d_out;

    const size_t Z_E  = (size_t)NB * ZH * ZH * NC;     // 3,981,312
    const size_t P_E  = (size_t)NB * PH * PH * NC;     // 14,204,928
    const size_t A192 = (size_t)NC  * KTOT;            // 921,600
    const size_t A576 = (size_t)MKC * KTOT;            // 2,764,800
    char* ws = (char*)d_ws;
    size_t off = 0;
    bf16_t* Z    = (bf16_t*)(ws + off); off += Z_E * 2;
    bf16_t* P1   = (bf16_t*)(ws + off); off += P_E * 2;
    bf16_t* P2   = (bf16_t*)(ws + off); off += P_E * 2;
    bf16_t* Ags1 = (bf16_t*)(ws + off); off += A192 * 2;
    bf16_t* Ags2 = (bf16_t*)(ws + off); off += A192 * 2;
    bf16_t* Ags3 = (bf16_t*)(ws + off); off += A576 * 2;
    bf16_t* Agm1 = (bf16_t*)(ws + off); off += A192 * 2;
    bf16_t* Agm2 = (bf16_t*)(ws + off); off += A192 * 2;
    bf16_t* Agm3 = (bf16_t*)(ws + off); off += A576 * 2;
    bf16_t* Agw1 = (bf16_t*)(ws + off); off += A192 * 2;
    bf16_t* Agw2 = (bf16_t*)(ws + off); off += A576 * 2;
    unsigned* Umax = (unsigned*)(ws + off); off += (size_t)NB * MKC * 4;

    hipMemsetAsync(P1, 0, P_E * 2, stream);
    hipMemsetAsync(P2, 0, P_E * 2, stream);
    hipMemsetAsync(Umax, 0, (size_t)NB * MKC * 4, stream);

    dim3 blk(256);
    pack_z_kernel<<<(unsigned)((Z_E + 255) / 256), blk, 0, stream>>>(z1, Z);
    pack_wdec_kernel<<<(unsigned)((A192 + 255) / 256), blk, 0, stream>>>(gs_w1, Ags1, NC, NC);
    pack_wdec_kernel<<<(unsigned)((A192 + 255) / 256), blk, 0, stream>>>(gs_w2, Ags2, NC, NC);
    pack_wconv_kernel<<<(unsigned)((A576 + 255) / 256), blk, 0, stream>>>(gs_w3, Ags3, NC, MKC);
    pack_wdec_kernel<<<(unsigned)((A192 + 255) / 256), blk, 0, stream>>>(gm_w1, Agm1, NC, NC);
    pack_wdec_kernel<<<(unsigned)((A192 + 255) / 256), blk, 0, stream>>>(gm_w2, Agm2, NC, NC);
    pack_wconv_kernel<<<(unsigned)((A576 + 255) / 256), blk, 0, stream>>>(gm_w3, Agm3, NC, MKC);
    pack_wdec_kernel<<<(unsigned)((A192 + 255) / 256), blk, 0, stream>>>(gw_w1, Agw1, NC, NC);
    pack_wdec_kernel<<<(unsigned)((A576 + 255) / 256), blk, 0, stream>>>(gw_w2, Agw2, NC, MKC);

    dim3 g1(8,  3, NB);   // 32x32 out, Cout=192
    dim3 g2(32, 3, NB);   // 64x64 out, Cout=192
    dim3 g3(32, 9, NB);   // 64x64 out, Cout=576

    // sigma: relu, relu, relu
    conv5_gemm<1, 0><<<g1, blk, 0, stream>>>(Z,  ZH, ZH, Ags1, gs_b1, NC,  5, P1, PH, PH);
    conv5_gemm<1, 1><<<g2, blk, 0, stream>>>(P1, PH, PH, Ags2, gs_b2, NC,  6, P2, PH, PH);
    conv5_gemm<1, 2><<<g3, blk, 0, stream>>>(P2, PH, PH, Ags3, gs_b3, MKC, 6, out, 0, 0);
    // means: leaky, leaky, none
    conv5_gemm<2, 0><<<g1, blk, 0, stream>>>(Z,  ZH, ZH, Agm1, gm_b1, NC,  5, P1, PH, PH);
    conv5_gemm<2, 1><<<g2, blk, 0, stream>>>(P1, PH, PH, Agm2, gm_b2, NC,  6, P2, PH, PH);
    conv5_gemm<0, 2><<<g3, blk, 0, stream>>>(P2, PH, PH, Agm3, gm_b3, MKC, 6, out + SIG_ELEMS, 0, 0);
    // weights: leaky, (deconv + global max fused), tail
    conv5_gemm<2, 0><<<g1, blk, 0, stream>>>(Z,  ZH, ZH, Agw1, gw_b1, NC,  5, P1, PH, PH);
    conv5_gemm<0, 3><<<g3, blk, 0, stream>>>(P1, PH, PH, Agw2, gw_b2, MKC, 6, Umax, 0, 0);
    weights_tail_kernel<<<NB, blk, 0, stream>>>(Umax, gw_w3, gw_b3, out + 2 * SIG_ELEMS);
}
